// HeadProbEncoder_85959475462781
// MI455X (gfx1250) — matrix operand order repaired and checked by reading, not compiled, not run on hardware
//
#include <hip/hip_runtime.h>
#include <hip/hip_bf16.h>
#include <math.h>
#include <stdint.h>

// ---------------------------------------------------------------------------
// HeadProbEncoder fused CDNA5 implementation.
// B=8, L=512, d=64, C=12 heads, K=2 distance groups, 4 iterations.
// Heavy contractions: v_wmma_f32_16x16x32_f16 (f16 in, f32 accum).
// A-query tiles staged into LDS with TENSOR_LOAD_TO_LDS (TDM).
// ---------------------------------------------------------------------------

typedef __attribute__((ext_vector_type(16))) _Float16 v16h;
typedef __attribute__((ext_vector_type(8)))  _Float16 v8h;
typedef __attribute__((ext_vector_type(8)))  float    v8f;
typedef __attribute__((ext_vector_type(4)))  unsigned int u32x4;
typedef __attribute__((ext_vector_type(8)))  int          i32x8;
typedef __attribute__((ext_vector_type(4)))  int          i32x4;

#define BB 8
#define LL 512
#define DD 64
#define CC 12
#define NROWS (BB*LL)            // 4096
#define NF    (2*CC*DD)          // 1536 features (k,c,b)
#define TSZ   (2*DD*DD*CC)       // 98304 ternary elements

static __device__ inline v8f wmma_f16(v16h a, v16h b, v8f c) {
  return __builtin_amdgcn_wmma_f32_16x16x32_f16(
      false, a, false, b, (short)0, c, false, false);
}

// A-matrix fragment (16x32 f16, M x K). base: row-major, row stride rs elems.
// ISA layout: lane m<16 holds K = kk+{0..7} (V0-3) and kk+{16..23} (V4-7);
// lane m+16 holds kk+{8..15} and kk+{24..31}.
static __device__ inline v16h load_a_frag(const _Float16* base, int rs,
                                          int row0, int kk, int lane) {
  int m  = lane & 15;
  int hi = lane >> 4;
  const _Float16* p = base + (size_t)(row0 + m) * rs + kk + hi * 8;
  v8h lo  = *(const v8h*)p;          // K = kk+hi*8 + 0..7
  v8h hi8 = *(const v8h*)(p + 16);   // K = kk+hi*8 + 16..23
  return __builtin_shufflevector(lo, hi8, 0,1,2,3,4,5,6,7,
                                          8,9,10,11,12,13,14,15);
}

// B-matrix fragment (32x16 f16, K x N) from Btranspose stored row-major
// [N][K] with row stride rs. Lane n<16 holds K kk+{0..7} and kk+{16..23},
// lane n+16 holds kk+{8..15} and kk+{24..31} -> two 16-byte vector loads.
static __device__ inline v16h load_b_frag(const _Float16* bT, int rs,
                                          int n0, int kk, int lane) {
  int n  = lane & 15;
  int hi = lane >> 4;
  const _Float16* p = bT + (size_t)(n0 + n) * rs + kk + hi * 8;
  v8h lo  = *(const v8h*)p;
  v8h hi8 = *(const v8h*)(p + 16);
  return __builtin_shufflevector(lo, hi8, 0,1,2,3,4,5,6,7,
                                          8,9,10,11,12,13,14,15);
}

// ---------------------------------------------------------------------------
// TDM: 2D tile load (16 rows x 64 f16 cols) from a row-major [512][64] f16
// tensor into LDS.  D# built per CDNA5 ISA ch.8 (groups 0/1; 2D -> g2/g3 = 0).
// ---------------------------------------------------------------------------
static __device__ inline void tdm_load_tile_16x64(const _Float16* gtile,
                                                  uint32_t lds_byte_off) {
  uint64_t ga = (uint64_t)(uintptr_t)gtile;
  u32x4 g0;
  g0[0] = 1u;                                   // count=1 (valid user D#)
  g0[1] = lds_byte_off;                         // lds_addr (bytes)
  g0[2] = (uint32_t)(ga & 0xFFFFFFFFu);         // global_addr[31:0]
  g0[3] = (uint32_t)((ga >> 32) & 0x01FFFFFFu)  // global_addr[56:32]
          | (2u << 30);                         // type = 2 ("image")
  i32x8 g1;
  g1[0] = (int)(1u << 16);        // workgroup_mask=0, data_size=1 (2 bytes)
  g1[1] = (int)(64u << 16);       // tensor_dim0 = 64 (bits 79:48, low part)
  g1[2] = (int)(512u << 16);      // tensor_dim0 hi=0 ; tensor_dim1 = 512 (low)
  g1[3] = (int)(64u << 16);       // tensor_dim1 hi=0 ; tile_dim0 = 64
  g1[4] = 16;                     // tile_dim1 = 16 ; tile_dim2 = 0
  g1[5] = 64;                     // tensor_dim0_stride = 64 (low 32)
  g1[6] = 0;                      // stride0 hi ; tensor_dim1_stride low
  g1[7] = 0;
  i32x4 gz = {0, 0, 0, 0};
#if defined(__clang_major__) && (__clang_major__ >= 23)
  i32x8 gz8 = {0, 0, 0, 0, 0, 0, 0, 0};
  __builtin_amdgcn_tensor_load_to_lds(g0, g1, gz, gz, gz8, 0);
#else
  __builtin_amdgcn_tensor_load_to_lds(g0, g1, gz, gz, 0);
#endif
}

// ---------------------------------------------------------------------------
// K0: pack T = 64*ternary into two f16 layouts:
//   Tq[k][c][b][a]  (Btranspose for the A-GEMM, rs=64)
//   Tg[a][f], f=(k*12+c)*64+b  (Btranspose for the G-GEMM, rs=1536)
// ---------------------------------------------------------------------------
__global__ __launch_bounds__(256)
void k_packT(const float* __restrict__ ternary,
             _Float16* __restrict__ Tq, _Float16* __restrict__ Tg) {
  int idx = blockIdx.x * 256 + threadIdx.x;
  if (idx >= TSZ) return;
  int c  = idx % CC;
  int t  = idx / CC;
  int b  = t % DD;
  int t2 = t / DD;
  int a  = t2 % DD;
  int k  = t2 / DD;
  float v = 64.0f * ternary[idx];           // idx == ((k*64+a)*64+b)*12+c
  int kc = k * CC + c;
  Tq[((size_t)kc * DD + b) * DD + a] = (_Float16)v;
  Tg[(size_t)a * NF + kc * DD + b]   = (_Float16)v;
}

// ---------------------------------------------------------------------------
// K1: row softmax over d=64 (+ mask1d), produce qz_h [row][64] f16 and
// qzT [z][a][i] f16 (per-z 64x512 transpose, PV values).
// ---------------------------------------------------------------------------
__global__ __launch_bounds__(128)
void k_softmax(const float* __restrict__ qz, const int* __restrict__ mask,
               _Float16* __restrict__ qzh, _Float16* __restrict__ qzT) {
  int row = blockIdx.x * 128 + threadIdx.x;
  if (row >= NROWS) return;
  int z = row >> 9, i = row & 511;
  const float* p = qz + (size_t)row * DD;
  float mx = -3.0e38f;
#pragma unroll
  for (int a = 0; a < DD; ++a) mx = fmaxf(mx, p[a]);
  float e[DD];
  float s = 0.f;
#pragma unroll
  for (int a = 0; a < DD; ++a) { e[a] = __expf(p[a] - mx); s += e[a]; }
  float inv = (mask[row] != 0) ? (1.0f / s) : 0.0f;
  _Float16* q = qzh + (size_t)row * DD;
  _Float16* t = qzT + ((size_t)z * DD) * LL + i;
#pragma unroll
  for (int a = 0; a < DD; ++a) {
    _Float16 v = (_Float16)(e[a] * inv);
    q[a] = v;
    t[(size_t)a * LL] = v;
  }
}

// ---------------------------------------------------------------------------
// K2: A_h[z][c][k][i][b] = sum_a qz[z,i,a] * T[k,a,b,c]
// grid.x = 256 row tiles, grid.y = 24 (k*12+c), one wave each.
// ---------------------------------------------------------------------------
__global__ __launch_bounds__(32)
void k_aqgemm(const _Float16* __restrict__ qzh,
              const _Float16* __restrict__ Tq,
              _Float16* __restrict__ Ah) {
  int lane = threadIdx.x;
  int row0 = blockIdx.x * 16;
  int kc   = blockIdx.y;                 // k*12 + c
  int z = row0 >> 9, i0 = row0 & 511;

  v16h a0 = load_a_frag(qzh, DD, row0, 0, lane);
  v16h a1 = load_a_frag(qzh, DD, row0, 32, lane);
  const _Float16* bT = Tq + (size_t)kc * DD * DD;

  _Float16* dst = Ah + (((size_t)z * CC + (kc % CC)) * 2 + (kc / CC)) * LL * DD
                     + (size_t)i0 * DD;
  int n = lane & 15, mo = (lane >> 4) * 8;
#pragma unroll
  for (int nt = 0; nt < 4; ++nt) {
    v8f acc = {};
    acc = wmma_f16(a0, load_b_frag(bT, DD, nt * 16, 0, lane), acc);
    acc = wmma_f16(a1, load_b_frag(bT, DD, nt * 16, 32, lane), acc);
#pragma unroll
    for (int r = 0; r < 8; ++r)
      dst[(size_t)(mo + r) * DD + nt * 16 + n] = (_Float16)acc[r];
  }
}

// ---------------------------------------------------------------------------
// K3: fused attention per (z, head c, 16-row i-block):
//   S[i,j] = A_{k(i,j)}[i,:]·qz[j,:], S[i,i]=0 ; softmax over j ; P[i,i]=0 ;
//   O_up = sum_{j>i} P qz[j],  O_lo = sum_{j<i} P qz[j]
// One wave per block.  A tiles via TDM->LDS, scores staged in LDS,
// P fragments rebuilt from S in registers at PV time.
// ---------------------------------------------------------------------------
#define SP 516   // f32 score row stride (padded, 16B-multiple)

__global__ __launch_bounds__(32)
void k_attn(const _Float16* __restrict__ Ah,
            const _Float16* __restrict__ qzh,
            const _Float16* __restrict__ qzT,
            _Float16* __restrict__ Oh) {
  __shared__ __align__(32) float    Ssh[16 * SP];
  __shared__ __align__(32) _Float16 Ash[2 * 16 * DD];   // A0 tile | A1 tile
  __shared__ float mstat[16];
  __shared__ float istat[16];

  int lane = threadIdx.x;
  int ib = blockIdx.x & 31;
  int c  = (blockIdx.x >> 5) % CC;
  int z  = blockIdx.x / (32 * CC);
  int i0 = ib * 16;

  // ---- TDM: stage the two 16x64 query tiles into LDS ----
  const _Float16* A0 = Ah + (((size_t)z * CC + c) * 2 + 0) * LL * DD
                          + (size_t)i0 * DD;
  const _Float16* A1 = A0 + (size_t)LL * DD;
  uint32_t ash_off = (uint32_t)(uintptr_t)(void*)Ash;   // flat->LDS offset
  tdm_load_tile_16x64(A0, ash_off);
  tdm_load_tile_16x64(A1, ash_off + 16 * DD * 2);
  __builtin_amdgcn_s_wait_tensorcnt(0);
  __syncthreads();

  v16h aq0k0 = load_a_frag(Ash, DD, 0, 0, lane);
  v16h aq0k1 = load_a_frag(Ash, DD, 0, 32, lane);
  v16h aq1k0 = load_a_frag(Ash + 16 * DD, DD, 0, 0, lane);
  v16h aq1k1 = load_a_frag(Ash + 16 * DD, DD, 0, 32, lane);

  const _Float16* Kz = qzh + (size_t)z * LL * DD;   // keys, Bt = [j][b]
  int n = lane & 15, mo = (lane >> 4) * 8;

  // ---- scores ----
  for (int jt = 0; jt < 32; ++jt) {
    int j0 = jt * 16;
    int cls = (j0 > i0 + 15) ? 0 : ((j0 + 15 < i0) ? 1 : 2);  // uniform
    v16h bk0 = load_b_frag(Kz, DD, j0, 0, lane);
    v16h bk1 = load_b_frag(Kz, DD, j0, 32, lane);
    v8f s0 = {}, s1 = {};
    if (cls != 1) { s0 = wmma_f16(aq0k0, bk0, s0); s0 = wmma_f16(aq0k1, bk1, s0); }
    if (cls != 0) { s1 = wmma_f16(aq1k0, bk0, s1); s1 = wmma_f16(aq1k1, bk1, s1); }
#pragma unroll
    for (int r = 0; r < 8; ++r) {
      int il = mo + r;
      int ig = i0 + il, jg = j0 + n;
      float v;
      if (cls == 0)      v = s0[r];
      else if (cls == 1) v = s1[r];
      else               v = (jg > ig) ? s0[r] : ((jg < ig) ? s1[r] : 0.0f);
      Ssh[il * SP + j0 + n] = v;
    }
  }
  __syncthreads();

  // ---- softmax stats over j (2 lanes per row, 256 cols each) ----
  {
    int row = lane & 15, hf = lane >> 4;
    const float* srow = Ssh + row * SP + hf * 256;
    float mx = -3.0e38f;
    for (int t = 0; t < 256; ++t) mx = fmaxf(mx, srow[t]);
    mx = fmaxf(mx, __shfl_xor(mx, 16, 32));
    float sm = 0.f;
    for (int t = 0; t < 256; ++t) sm += __expf(srow[t] - mx);
    sm += __shfl_xor(sm, 16, 32);
    if (hf == 0) { mstat[row] = mx; istat[row] = 1.0f / sm; }
  }
  __syncthreads();

  // ---- PV: rebuild P fragments from S, split upper/lower accumulation ----
  const _Float16* Vt = qzT + (size_t)z * DD * LL;   // Bt = [b][j]
  v8f oup[4] = {}, olo[4] = {};
  {
    int m = lane & 15, hi = lane >> 4;
    float mx  = mstat[m];
    float inv = istat[m];
    int ig = i0 + m;
    for (int jc = 0; jc < 16; ++jc) {
      int j0 = jc * 32;
      int cls = (j0 > i0 + 15) ? 0 : ((j0 + 31 < i0) ? 1 : 2);  // uniform
      const float* srow = Ssh + m * SP + j0 + hi * 8;
      v16h pu, pl;
#pragma unroll
      for (int e = 0; e < 16; ++e) {
        int jloc = (e < 8) ? e : (8 + e);          // {0..7, 16..23}
        int jg = j0 + hi * 8 + jloc;
        float pv = __expf(srow[jloc] - mx) * inv;  // P[i,j]; diag masked below
        pu[e] = (jg > ig) ? (_Float16)pv : (_Float16)0;
        pl[e] = (jg < ig) ? (_Float16)pv : (_Float16)0;
      }
#pragma unroll
      for (int nt = 0; nt < 4; ++nt) {
        v16h bf = load_b_frag(Vt, LL, nt * 16, j0, lane);
        if (cls != 1) oup[nt] = wmma_f16(pu, bf, oup[nt]);
        if (cls != 0) olo[nt] = wmma_f16(pl, bf, olo[nt]);
      }
    }
  }

  // ---- store O into feature tensor [row][f], f = k*768 + c*64 + b ----
  _Float16* obase = Oh + (size_t)(z * LL + i0) * NF;
#pragma unroll
  for (int nt = 0; nt < 4; ++nt) {
#pragma unroll
    for (int r = 0; r < 8; ++r) {
      size_t roff = (size_t)(mo + r) * NF + nt * 16 + n;
      obase[roff + c * DD]       = (_Float16)oup[nt][r];   // k=0
      obase[roff + 768 + c * DD] = (_Float16)olo[nt][r];   // k=1
    }
  }
}

// ---------------------------------------------------------------------------
// K4: G[row,a] = sum_f O[row,f]*Tg[a,f] ; q_z = (x + G)/64
// ---------------------------------------------------------------------------
__global__ __launch_bounds__(32)
void k_ggemm(const _Float16* __restrict__ Oh, const _Float16* __restrict__ Tg,
             const float* __restrict__ x, float* __restrict__ qz) {
  int lane = threadIdx.x;
  int row0 = blockIdx.x * 16;
  const _Float16* Arow = Oh + (size_t)row0 * NF;
  v8f acc[4] = {};
  for (int kk = 0; kk < NF; kk += 32) {
    if (kk + 32 < NF)   // CDNA5 global_prefetch_b8 for next A chunk
      __builtin_prefetch((const void*)(Arow + (size_t)(lane & 15) * NF + kk + 32), 0, 1);
    v16h af = load_a_frag(Arow, NF, 0, kk, lane);
#pragma unroll
    for (int nt = 0; nt < 4; ++nt)
      acc[nt] = wmma_f16(af, load_b_frag(Tg, NF, nt * 16, kk, lane), acc[nt]);
  }
  int n = lane & 15, mo = (lane >> 4) * 8;
#pragma unroll
  for (int nt = 0; nt < 4; ++nt)
#pragma unroll
    for (int r = 0; r < 8; ++r) {
      size_t idx = (size_t)(row0 + mo + r) * DD + nt * 16 + n;
      qz[idx] = (x[idx] + acc[nt][r]) * 0.015625f;   // /64
    }
}

// ---------------------------------------------------------------------------
extern "C" void kernel_launch(void* const* d_in, const int* in_sizes, int n_in,
                              void* d_out, int out_size, void* d_ws, size_t ws_size,
                              hipStream_t stream) {
  (void)in_sizes; (void)n_in; (void)out_size; (void)ws_size;
  const float* x       = (const float*)d_in[0];   // [8,512,64]
  const int*   mask    = (const int*)d_in[1];     // [8,512]
  const float* ternary = (const float*)d_in[2];   // [2,64,64,12]
  float* out = (float*)d_out;

  char* ws = (char*)d_ws;
  size_t off = 0;
  auto carve = [&](size_t bytes) -> char* {
    char* p = ws + off;
    off = (off + bytes + 255) & ~(size_t)255;
    return p;
  };
  float*     qz  = (float*)    carve((size_t)NROWS * DD * 4);   // 1 MB
  _Float16*  qzh = (_Float16*) carve((size_t)NROWS * DD * 2);
  _Float16*  qzT = (_Float16*) carve((size_t)NROWS * DD * 2);
  _Float16*  Tq  = (_Float16*) carve((size_t)TSZ * 2);
  _Float16*  Tg  = (_Float16*) carve((size_t)TSZ * 2);
  _Float16*  Ah  = (_Float16*) carve((size_t)NROWS * NF * 2);   // 12 MB
  _Float16*  Oh  = (_Float16*) carve((size_t)NROWS * NF * 2);   // 12 MB

  // q_z(0) = x
  hipMemcpyAsync(qz, x, (size_t)NROWS * DD * 4, hipMemcpyDeviceToDevice, stream);
  k_packT<<<(TSZ + 255) / 256, 256, 0, stream>>>(ternary, Tq, Tg);

  for (int it = 0; it < 4; ++it) {
    k_softmax<<<NROWS / 128, 128, 0, stream>>>(qz, mask, qzh, qzT);
    k_aqgemm<<<dim3(NROWS / 16, 2 * CC), 32, 0, stream>>>(qzh, Tq, Ah);
    k_attn<<<BB * CC * (LL / 16), 32, 0, stream>>>(Ah, qzh, qzT, Oh);
    k_ggemm<<<NROWS / 16, 32, 0, stream>>>(Oh, Tg, x, qz);
  }

  hipMemcpyAsync(out, qz, (size_t)NROWS * DD * 4, hipMemcpyDeviceToDevice, stream);
}
